// EngagementPredictor_72980084293983
// MI455X (gfx1250) — matrix operand order repaired and checked by reading, not compiled, not run on hardware
//
#include <hip/hip_runtime.h>
#include <math.h>

typedef __bf16 bf16;
typedef __attribute__((ext_vector_type(16))) __bf16 bf16x16;
typedef __attribute__((ext_vector_type(8)))  __bf16 bf16x8;
typedef __attribute__((ext_vector_type(8)))  float  f32x8;

// ---------------------------------------------------------------------------
// WMMA fragment helpers (wave32, 16x16x32 bf16, layouts per CDNA5 ISA 7.12.2)
// ---------------------------------------------------------------------------
static __device__ __forceinline__ bf16x16 frag_zero() {
  bf16x16 f;
#pragma unroll
  for (int i = 0; i < 16; ++i) f[i] = (bf16)0.0f;
  return f;
}

// A fragment: 16(M) x 32(K) tile, row-major source, src points at tile row 0.
// lane<16: K 0..7 (v0-3) and 16..23 (v4-7); lane>=16: K 8..15 and 24..31.
template <bool GUARD>
static __device__ __forceinline__ bf16x16 frag_a_load(const bf16* src, int lda,
                                                      int k0, int mvalid) {
  const int lane = threadIdx.x & 31;
  const int r = lane & 15, hi = lane >> 4;
  if (GUARD && r >= mvalid) return frag_zero();
  const bf16* p = src + (size_t)r * lda + k0 + hi * 8;
  bf16x8 lo  = *(const bf16x8*)p;
  bf16x8 hi8 = *(const bf16x8*)(p + 16);
  bf16x16 f;
#pragma unroll
  for (int i = 0; i < 8; ++i) { f[i] = lo[i]; f[8 + i] = hi8[i]; }
  return f;
}

// B fragment: 32(K) x 16(N) operand, stored transposed (N rows, K contiguous).
// lane<16: col=lane, K 0..7 and 16..23; lane>=16: col=lane-16, K 8..15 and 24..31.
template <bool GUARD>
static __device__ __forceinline__ bf16x16 frag_b_load(const bf16* srcT, int ldb,
                                                      int k0, int nvalid) {
  const int lane = threadIdx.x & 31;
  const int c = lane & 15, hi = lane >> 4;
  if (GUARD && c >= nvalid) return frag_zero();
  const bf16* p = srcT + (size_t)c * ldb + k0 + hi * 8;
  bf16x8 a = *(const bf16x8*)p;
  bf16x8 b = *(const bf16x8*)(p + 16);
  bf16x16 f;
#pragma unroll
  for (int i = 0; i < 8; ++i) { f[i] = a[i]; f[8 + i] = b[i]; }
  return f;
}

static __device__ __forceinline__ f32x8 wmma_bf16(bf16x16 a, bf16x16 b, f32x8 c) {
  return __builtin_amdgcn_wmma_f32_16x16x32_bf16(false, a, false, b,
                                                 (short)0, c, false, false);
}

// ---------------------------------------------------------------------------
// Generic WMMA GEMM:  C[M,N] = A[M,K](bf16) * Bt[N,K](bf16)^T + bias
// OUTMODE: 0 = f32 row-major, 1 = bf16 row-major, 2 = bf16 transposed [N,M]
// One wave = (MT*16) x 64 output tile; k-loop unrolled x2 with two explicit
// fragment buffer sets (software pipeline; requires K % 64 == 0).
// ---------------------------------------------------------------------------
template <int OUTMODE, bool RELU, bool GUARD, int MT>
__global__ __launch_bounds__(128) void gemm_wmma(
    const bf16* __restrict__ A, const bf16* __restrict__ Bt,
    const float* __restrict__ bias, void* __restrict__ Cout,
    int M, int N, int K, int ldc) {
  const int wave = threadIdx.x >> 5;
  const int lane = threadIdx.x & 31;
  const int c15 = lane & 15, hi = lane >> 4;
  const int row0 = (blockIdx.y * 4 + wave) * (16 * MT);
  const int col0 = blockIdx.x * 64;
  if (GUARD && row0 >= M) return;

  int mv[MT], nv[4];
#pragma unroll
  for (int mt = 0; mt < MT; ++mt) {
    const int rem = M - (row0 + mt * 16);
    mv[mt] = rem < 16 ? (rem < 0 ? 0 : rem) : 16;
  }
#pragma unroll
  for (int j = 0; j < 4; ++j) nv[j] = N - (col0 + j * 16);

  f32x8 acc[MT][4];
#pragma unroll
  for (int mt = 0; mt < MT; ++mt)
#pragma unroll
    for (int j = 0; j < 4; ++j)
#pragma unroll
      for (int r = 0; r < 8; ++r) acc[mt][j][r] = 0.0f;

  const bf16* Arow = A + (size_t)row0 * K;

  bf16x16 a0[MT], b0[4], a1[MT], b1[4];
#pragma unroll
  for (int mt = 0; mt < MT; ++mt)
    a0[mt] = frag_a_load<GUARD>(Arow + (size_t)mt * 16 * K, K, 0, mv[mt]);
#pragma unroll
  for (int j = 0; j < 4; ++j)
    b0[j] = frag_b_load<GUARD>(Bt + (size_t)(col0 + j * 16) * K, K, 0, nv[j]);

  for (int k0 = 0; k0 < K; k0 += 64) {
    // stage 1: prefetch k0+32, compute on (a0,b0)
#pragma unroll
    for (int mt = 0; mt < MT; ++mt)
      a1[mt] = frag_a_load<GUARD>(Arow + (size_t)mt * 16 * K, K, k0 + 32, mv[mt]);
#pragma unroll
    for (int j = 0; j < 4; ++j)
      b1[j] = frag_b_load<GUARD>(Bt + (size_t)(col0 + j * 16) * K, K, k0 + 32, nv[j]);
#pragma unroll
    for (int j = 0; j < 4; ++j)
#pragma unroll
      for (int mt = 0; mt < MT; ++mt)
        acc[mt][j] = wmma_bf16(a0[mt], b0[j], acc[mt][j]);

    // stage 2: prefetch k0+64 (wrap to 0 on last iter; always in-bounds),
    // compute on (a1,b1)
    const int kn = (k0 + 64 < K) ? (k0 + 64) : 0;
#pragma unroll
    for (int mt = 0; mt < MT; ++mt)
      a0[mt] = frag_a_load<GUARD>(Arow + (size_t)mt * 16 * K, K, kn, mv[mt]);
#pragma unroll
    for (int j = 0; j < 4; ++j)
      b0[j] = frag_b_load<GUARD>(Bt + (size_t)(col0 + j * 16) * K, K, kn, nv[j]);
#pragma unroll
    for (int j = 0; j < 4; ++j)
#pragma unroll
      for (int mt = 0; mt < MT; ++mt)
        acc[mt][j] = wmma_bf16(a1[mt], b1[j], acc[mt][j]);
  }

#pragma unroll
  for (int mt = 0; mt < MT; ++mt)
#pragma unroll
    for (int j = 0; j < 4; ++j) {
      const int n = col0 + j * 16 + c15;
      if (GUARD && n >= N) continue;
      const float bv = bias ? bias[n] : 0.0f;
#pragma unroll
      for (int r = 0; r < 8; ++r) {
        const int row = row0 + mt * 16 + r + 8 * hi;
        if (GUARD && row >= M) continue;
        float v = acc[mt][j][r] + bv;
        if (RELU) v = fmaxf(v, 0.0f);
        if (OUTMODE == 0)      ((float*)Cout)[(size_t)row * ldc + n] = v;
        else if (OUTMODE == 1) ((bf16*)Cout)[(size_t)row * ldc + n] = (bf16)v;
        else                   ((bf16*)Cout)[(size_t)n * ldc + row] = (bf16)v;
      }
    }
}

// ---------------------------------------------------------------------------
// Flash attention. One wave per (batch, head, 16-query tile).
// Q:[B*S,1024] bf16, Km:[B*S,1024] bf16 (keys are natural B operands),
// Vt:[1024, B*S] bf16 (produced by transposed-store V projection).
// ---------------------------------------------------------------------------
template <int D>
__global__ __launch_bounds__(32) void flash_attn(
    const bf16* __restrict__ Q, const bf16* __restrict__ Km,
    const bf16* __restrict__ Vt, const int* __restrict__ mask,
    bf16* __restrict__ Octx, int S, int BS) {
  const int q0 = blockIdx.x * 16;
  const int h  = blockIdx.y;
  const int b  = blockIdx.z;
  const int lane = threadIdx.x & 31;
  const int c15 = lane & 15, hi = lane >> 4;
  const float scale = 1.0f / sqrtf((float)D);

  __shared__ bf16 qs[16 * D];
  __shared__ bf16 ps[16 * 32];

  // Stage Q tile (16 x D) into LDS, then keep fragments in registers.
  const bf16* qsrc = Q + (size_t)(b * S + q0) * 1024 + h * D;
  for (int idx = lane; idx < 16 * (D / 8); idx += 32) {
    const int row = idx / (D / 8), col8 = idx % (D / 8);
    *(bf16x8*)&qs[row * D + col8 * 8] =
        *(const bf16x8*)&qsrc[(size_t)row * 1024 + col8 * 8];
  }
  __syncthreads();

  bf16x16 qf[D / 32];
#pragma unroll
  for (int kk = 0; kk < D / 32; ++kk)
    qf[kk] = frag_a_load<false>(qs, D, kk * 32, 16);

  f32x8 acc[D / 16];
#pragma unroll
  for (int j = 0; j < D / 16; ++j)
#pragma unroll
    for (int r = 0; r < 8; ++r) acc[j][r] = 0.0f;

  float mrow[8], lrow[8];
#pragma unroll
  for (int r = 0; r < 8; ++r) { mrow[r] = -__builtin_huge_valf(); lrow[r] = 0.0f; }

  const int* mp = mask + b * S;

  for (int k0 = 0; k0 < S; k0 += 32) {
    // scores: two 16x16 key tiles, K-accumulate over head dim
    f32x8 s[2];
#pragma unroll
    for (int nt = 0; nt < 2; ++nt) {
#pragma unroll
      for (int r = 0; r < 8; ++r) s[nt][r] = 0.0f;
      const bf16* krow = Km + (size_t)(b * S + k0 + nt * 16) * 1024 + h * D;
#pragma unroll
      for (int kk = 0; kk < D / 32; ++kk)
        s[nt] = wmma_bf16(qf[kk], frag_b_load<false>(krow, 1024, kk * 32, 16), s[nt]);
      const int mv = mp[k0 + nt * 16 + c15];
#pragma unroll
      for (int r = 0; r < 8; ++r)
        s[nt][r] = mv ? s[nt][r] * scale : -1e9f;
    }
    // online softmax (rows r / r+8 live in lane halves; reduce across 16 lanes)
    float alpha[8];
#pragma unroll
    for (int r = 0; r < 8; ++r) {
      float cm = fmaxf(s[0][r], s[1][r]);
#pragma unroll
      for (int off = 1; off < 16; off <<= 1) cm = fmaxf(cm, __shfl_xor(cm, off, 32));
      const float mn = fmaxf(mrow[r], cm);
      alpha[r] = __expf(mrow[r] - mn);
      mrow[r] = mn;
      const float p0 = __expf(s[0][r] - mn);
      const float p1 = __expf(s[1][r] - mn);
      s[0][r] = p0; s[1][r] = p1;
      float rs = p0 + p1;
#pragma unroll
      for (int off = 1; off < 16; off <<= 1) rs += __shfl_xor(rs, off, 32);
      lrow[r] = lrow[r] * alpha[r] + rs;
    }
#pragma unroll
    for (int j = 0; j < D / 16; ++j)
#pragma unroll
      for (int r = 0; r < 8; ++r) acc[j][r] *= alpha[r];

    // C-layout P -> LDS -> A-layout fragment
#pragma unroll
    for (int nt = 0; nt < 2; ++nt)
#pragma unroll
      for (int r = 0; r < 8; ++r)
        ps[(r + 8 * hi) * 32 + nt * 16 + c15] = (bf16)s[nt][r];
    __syncthreads();
    bf16x16 pf = frag_a_load<false>(ps, 32, 0, 16);

    // ctx += P (16x32) * V (32xD); Vt rows are output dims, K contiguous
#pragma unroll
    for (int j = 0; j < D / 16; ++j) {
      const bf16* vrow = Vt + (size_t)(h * D + j * 16) * BS + b * S + k0;
      acc[j] = wmma_bf16(pf, frag_b_load<false>(vrow, BS, 0, 16), acc[j]);
    }
    __syncthreads();
  }

  bf16* orow = Octx + (size_t)(b * S + q0) * 1024 + h * D;
#pragma unroll
  for (int r = 0; r < 8; ++r) {
    const float inv = 1.0f / lrow[r];
#pragma unroll
    for (int j = 0; j < D / 16; ++j)
      orow[(size_t)(r + 8 * hi) * 1024 + j * 16 + c15] = (bf16)(acc[j][r] * inv);
  }
}

// ---------------------------------------------------------------------------
// Data-prep + pooling kernels
// ---------------------------------------------------------------------------
__global__ void f32_to_bf16_k(const float* __restrict__ in, bf16* __restrict__ out, int n) {
  const int i = blockIdx.x * blockDim.x + threadIdx.x;
  if (i < n) out[i] = (bf16)in[i];
}

// W[K,N] f32 -> Wt[N,K] bf16
__global__ void transpose_to_bf16_k(const float* __restrict__ W, bf16* __restrict__ Wt,
                                    int K, int N) {
  const int i = blockIdx.x * blockDim.x + threadIdx.x;
  if (i < K * N) {
    const int nIdx = i / K, k = i % K;
    Wt[(size_t)nIdx * K + k] = (bf16)W[(size_t)k * N + nIdx];
  }
}

// masked mean pool: fused[b, which*H + n] = sum_s X[b*S+s, n]*mask / sum_s mask
__global__ void masked_pool_k(const bf16* __restrict__ X, const int* __restrict__ mask,
                              bf16* __restrict__ fused, int which, int S, int H) {
  const int i = blockIdx.x * blockDim.x + threadIdx.x;
  const int b = i / H, n = i % H;
  if (i >= 8 * H) return;
  const int* mp = mask + b * S;
  float sum = 0.0f; int den = 0;
  for (int s = 0; s < S; ++s) {
    const int mv = mp[s];
    den += mv;
    sum += mv ? (float)X[(size_t)(b * S + s) * H + n] : 0.0f;
  }
  fused[(size_t)b * (3 * H) + which * H + n] = (bf16)(sum / (float)den);
}

// ---------------------------------------------------------------------------
// Host orchestration
// ---------------------------------------------------------------------------
extern "C" void kernel_launch(void* const* d_in, const int* in_sizes, int n_in,
                              void* d_out, int out_size, void* d_ws, size_t ws_size,
                              hipStream_t stream) {
  (void)in_sizes; (void)n_in; (void)out_size; (void)ws_size;
  constexpr int Bn = 8, S = 1024, H = 1024;
  constexpr int BS = Bn * S;

  const float* hidden = (const float*)d_in[0];
  const int*   mask   = (const int*)d_in[1];

  char* base = (char*)d_ws;
  size_t off = 0;
  auto alloc_bf16 = [&](size_t elems) -> bf16* {
    bf16* p = (bf16*)(base + off);
    off += ((elems * sizeof(bf16)) + 255) & ~(size_t)255;
    return p;
  };

  bf16* xb = alloc_bf16((size_t)BS * H);
  bf16* wt[12];
  for (int i = 0; i < 12; ++i) wt[i] = alloc_bf16((size_t)H * H);
  bf16* f1t = alloc_bf16((size_t)H * 3 * H);       // [1024, 3072]
  bf16* f2t = alloc_bf16((size_t)(H / 2) * H);     // [512, 1024]
  bf16* clt = alloc_bf16((size_t)2 * (H / 2));     // [2, 512]
  bf16* qb  = alloc_bf16((size_t)BS * H);
  bf16* kb  = alloc_bf16((size_t)BS * H);
  bf16* vt  = alloc_bf16((size_t)BS * H);          // transposed: [1024, BS]
  bf16* cx  = alloc_bf16((size_t)BS * H);
  bf16* ao  = alloc_bf16((size_t)BS * H);
  bf16* fu  = alloc_bf16((size_t)Bn * 3 * H);
  bf16* h1  = alloc_bf16((size_t)Bn * H);
  bf16* h2  = alloc_bf16((size_t)Bn * (H / 2));

  // Big tile-aligned GEMMs: unguarded, 32x64 per wave (MT=2).
  auto gemm_big = [&](int mode, const bf16* A, const bf16* Bt, const float* bias,
                      void* C, int M, int N, int K, int ldc) {
    dim3 g(N / 64, M / 128), blk(128);
    if (mode == 1)      gemm_wmma<1, false, false, 2><<<g, blk, 0, stream>>>(A, Bt, bias, C, M, N, K, ldc);
    else                gemm_wmma<2, false, false, 2><<<g, blk, 0, stream>>>(A, Bt, bias, C, M, N, K, ldc);
  };
  // Small guarded GEMMs (M=8, N down to 2): MT=1.
  auto gemm_small = [&](int mode, bool relu, const bf16* A, const bf16* Bt,
                        const float* bias, void* C, int M, int N, int K, int ldc) {
    dim3 g((N + 63) / 64, (M + 63) / 64), blk(128);
    if (mode == 0)      gemm_wmma<0, false, true, 1><<<g, blk, 0, stream>>>(A, Bt, bias, C, M, N, K, ldc);
    else if (relu)      gemm_wmma<1, true,  true, 1><<<g, blk, 0, stream>>>(A, Bt, bias, C, M, N, K, ldc);
    else                gemm_wmma<1, false, true, 1><<<g, blk, 0, stream>>>(A, Bt, bias, C, M, N, K, ldc);
  };

  // hidden -> bf16
  {
    const int n = BS * H;
    f32_to_bf16_k<<<dim3((n + 255) / 256), dim3(256), 0, stream>>>(hidden, xb, n);
  }
  // attention weights -> transposed bf16 (q,k,v,o per MHA)
  for (int m = 0; m < 3; ++m)
    for (int w = 0; w < 4; ++w) {
      const float* W = (const float*)d_in[2 + m * 8 + w * 2];
      const int n = H * H;
      transpose_to_bf16_k<<<dim3((n + 255) / 256), dim3(256), 0, stream>>>(
          W, wt[m * 4 + w], H, H);
    }
  { const int n = 3 * H * H;
    transpose_to_bf16_k<<<dim3((n + 255) / 256), dim3(256), 0, stream>>>(
        (const float*)d_in[26], f1t, 3 * H, H); }
  { const int n = H * (H / 2);
    transpose_to_bf16_k<<<dim3((n + 255) / 256), dim3(256), 0, stream>>>(
        (const float*)d_in[28], f2t, H, H / 2); }
  { const int n = (H / 2) * 2;
    transpose_to_bf16_k<<<dim3((n + 255) / 256), dim3(256), 0, stream>>>(
        (const float*)d_in[30], clt, H / 2, 2); }

  // three MHAs (beh: 8 heads d=128; tmp/pat: 4 heads d=256)
  for (int m = 0; m < 3; ++m) {
    const float* qbias = (const float*)d_in[2 + m * 8 + 1];
    const float* kbias = (const float*)d_in[2 + m * 8 + 3];
    const float* vbias = (const float*)d_in[2 + m * 8 + 5];
    const float* obias = (const float*)d_in[2 + m * 8 + 7];
    gemm_big(1, xb, wt[m * 4 + 0], qbias, qb, BS, H, H, H);
    gemm_big(1, xb, wt[m * 4 + 1], kbias, kb, BS, H, H, H);
    gemm_big(2, xb, wt[m * 4 + 2], vbias, vt, BS, H, H, BS);  // V^T
    if (m == 0)
      flash_attn<128><<<dim3(S / 16, 8, Bn), dim3(32), 0, stream>>>(qb, kb, vt, mask, cx, S, BS);
    else
      flash_attn<256><<<dim3(S / 16, 4, Bn), dim3(32), 0, stream>>>(qb, kb, vt, mask, cx, S, BS);
    gemm_big(1, cx, wt[m * 4 + 3], obias, ao, BS, H, H, H);
    masked_pool_k<<<dim3((Bn * H + 255) / 256), dim3(256), 0, stream>>>(ao, mask, fu, m, S, H);
  }

  // fusion MLP + classifier
  gemm_small(1, true,  fu, f1t, (const float*)d_in[27], h1, Bn, H, 3 * H, H);
  gemm_small(1, true,  h1, f2t, (const float*)d_in[29], h2, Bn, H / 2, H, H / 2);
  gemm_small(0, false, h2, clt, (const float*)d_in[31], d_out, Bn, 2, H / 2, 2);
}
